// FastWeightLayer_82652350644605
// MI455X (gfx1250) — compile-verified, matrix operand order checked by reading, not run on hardware
//
#include <hip/hip_runtime.h>
#include <hip/hip_bf16.h>
#include <math.h>

// ---------------------------------------------------------------------------
// Problem constants (from reference): B=4,S=512 -> Nt=2048; H=1024; F=512; V=32000
// ---------------------------------------------------------------------------
#define NT 2048
#define HD 1024
#define FD 512
#define VD 32000
// softmax stats chunking: 25 chunks of 1280 cols
#define NCH 25
#define CHC 1280
// backward P chunking: 10 chunks of 3200 cols, 320 cols per workgroup
#define NPB 10
#define CHB 3200
#define CPW 320

typedef unsigned short bf16_t;
typedef __attribute__((ext_vector_type(16))) __bf16 v16bf;
typedef __attribute__((ext_vector_type(8)))  float  v8f;

__device__ __forceinline__ bf16_t f2bf(float f) {
  unsigned u = __float_as_uint(f);
  u += 0x7FFFu + ((u >> 16) & 1u);           // round-to-nearest-even
  return (bf16_t)(u >> 16);
}

// A/B fragment loader for v_wmma_f32_16x16x32_bf16.
// Lane L holds 16 contiguous K elems of row (L%16), K offset (L/16)*16.
// Works for A [16xK rowmajor] and for B given as B^T rowmajor [16 x K].
__device__ __forceinline__ v16bf load_frag(const bf16_t* base, int ld) {
  int lane = threadIdx.x & 31;
  const bf16_t* p = base + (long)(lane & 15) * ld + ((lane >> 4) << 4);
  return *(const v16bf*)p;
}

// Same fragment layout, but from an LDS tile with fixed ld = 32 elements.
__device__ __forceinline__ v16bf lds_frag(const bf16_t* base) {
  int lane = threadIdx.x & 31;
  const bf16_t* p = base + (lane & 15) * 32 + ((lane >> 4) << 4);
  return *(const v16bf*)p;
}

__device__ __forceinline__ v8f wmma_bf16(v16bf a, v16bf b, v8f c) {
  return __builtin_amdgcn_wmma_f32_16x16x32_bf16(false, a, false, b, (short)0, c,
                                                 false, false);
}

// CDNA5 async copy: global -> LDS, 16 bytes per lane, tracked by ASYNCcnt.
// (CDNA5 ISA; inline asm since no portable builtin.)
__device__ __forceinline__ void async_copy_b128(unsigned lds_off,
                                                const bf16_t* gptr) {
  asm volatile("global_load_async_to_lds_b128 %0, %1, off"
               :: "v"(lds_off), "v"(gptr)
               : "memory");
}
__device__ __forceinline__ void wait_async_le1() {
  asm volatile("s_wait_asynccnt 0x1" ::: "memory");
}
__device__ __forceinline__ void wait_async_le0() {
  asm volatile("s_wait_asynccnt 0x0" ::: "memory");
}

// ---------------------------------------------------------------------------
// Generic bf16 GEMM: C[M,N] = A[M,K] @ B[N,K]^T  (+bias1+bias2, relu, acc)
// block = 256 threads = 8 waves (4 M x 2 N), wave tile 32x32, wg tile 128x64.
// B tile (64x32 bf16 per K-step) is staged through LDS with double-buffered
// global_load_async_to_lds_b128 and shared by all 8 waves.
// M%128==0, N%64==0, K%32==0 required.
// ---------------------------------------------------------------------------
__global__ __launch_bounds__(256) void gemm_bt_kernel(
    const bf16_t* __restrict__ A, int lda,
    const bf16_t* __restrict__ B, int ldb,
    float* __restrict__ C, int ldc, int K,
    const float* __restrict__ bias1, const float* __restrict__ bias2,
    int relu, int acc, bf16_t* __restrict__ Cbf, int ldcb) {
  __shared__ __attribute__((aligned(32))) bf16_t smB[2][64 * 32];  // 2 x 4KB

  int tid = threadIdx.x;
  int wid = tid >> 5;
  int wm = wid & 3, wn = wid >> 2;
  int mt = blockIdx.x * 128 + wm * 32;
  int lane = tid & 31;
  int half = lane >> 4, cc = lane & 15;
  int ntLocal = wn * 32;
  int nt = blockIdx.y * 64 + ntLocal;

  v8f c00{}, c01{}, c10{}, c11{};
  if (acc) {
    for (int r = 0; r < 8; ++r) {
      int r0 = mt + r + half * 8, r1 = r0 + 16;
      c00[r] = C[(long)r0 * ldc + nt + cc];
      c01[r] = C[(long)r0 * ldc + nt + 16 + cc];
      c10[r] = C[(long)r1 * ldc + nt + cc];
      c11[r] = C[(long)r1 * ldc + nt + 16 + cc];
    }
  }
  const bf16_t* A0 = A + (long)mt * lda;
  const bf16_t* A1 = A + (long)(mt + 16) * lda;

  // async-copy mapping: thread t moves 8 bf16 (one b128):
  // local B row = t/4 (0..63), 8-elem segment = t%4.
  int crow = tid >> 2;
  int cseg = (tid & 3) << 3;
  const bf16_t* Bsrc = B + (long)(blockIdx.y * 64 + crow) * ldb + cseg;
  unsigned ldsDst0 = (unsigned)(size_t)&smB[0][crow * 32 + cseg];
  unsigned ldsDst1 = (unsigned)(size_t)&smB[1][crow * 32 + cseg];

  async_copy_b128(ldsDst0, Bsrc);  // prime buffer 0 (k = 0)

  int nsteps = K >> 5;
  for (int i = 0; i < nsteps; ++i) {
    int k = i << 5;
    if (i + 1 < nsteps) {
      async_copy_b128(((i + 1) & 1) ? ldsDst1 : ldsDst0, Bsrc + k + 32);
      wait_async_le1();            // wait only for the older (current) buffer
    } else {
      wait_async_le0();
    }
    __syncthreads();               // current buffer visible to all waves

    __builtin_prefetch(A0 + k + 128, 0, 0);
    const bf16_t* bufp = smB[i & 1];
    v16bf a0 = load_frag(A0 + k, lda);
    v16bf a1 = load_frag(A1 + k, lda);
    v16bf b0 = lds_frag(bufp + ntLocal * 32);
    v16bf b1 = lds_frag(bufp + (ntLocal + 16) * 32);
    c00 = wmma_bf16(a0, b0, c00);
    c01 = wmma_bf16(a0, b1, c01);
    c10 = wmma_bf16(a1, b0, c10);
    c11 = wmma_bf16(a1, b1, c11);

    __syncthreads();               // done reading: buffer may be re-filled
  }

  float bA0 = 0.f, bA1 = 0.f;
  if (bias1) { bA0 += bias1[nt + cc]; bA1 += bias1[nt + 16 + cc]; }
  if (bias2) { bA0 += bias2[nt + cc]; bA1 += bias2[nt + 16 + cc]; }
  for (int r = 0; r < 8; ++r) {
    int r0 = mt + r + half * 8, r1 = r0 + 16;
    float v00 = c00[r] + bA0, v01 = c01[r] + bA1;
    float v10 = c10[r] + bA0, v11 = c11[r] + bA1;
    if (relu) {
      v00 = fmaxf(v00, 0.f); v01 = fmaxf(v01, 0.f);
      v10 = fmaxf(v10, 0.f); v11 = fmaxf(v11, 0.f);
    }
    C[(long)r0 * ldc + nt + cc]      = v00;
    C[(long)r0 * ldc + nt + 16 + cc] = v01;
    C[(long)r1 * ldc + nt + cc]      = v10;
    C[(long)r1 * ldc + nt + 16 + cc] = v11;
    if (Cbf) {
      Cbf[(long)r0 * ldcb + nt + cc]      = f2bf(v00);
      Cbf[(long)r0 * ldcb + nt + 16 + cc] = f2bf(v01);
      Cbf[(long)r1 * ldcb + nt + cc]      = f2bf(v10);
      Cbf[(long)r1 * ldcb + nt + 16 + cc] = f2bf(v11);
    }
  }
}

// ---------------------------------------------------------------------------
// Fused logits + online-softmax partial stats over a V-chunk.
// grid = (NT/64, NCH); block = 256 (waves 4M x 2N). A frags cached in VGPRs.
// ---------------------------------------------------------------------------
__global__ __launch_bounds__(256) void logits_stats_kernel(
    const bf16_t* __restrict__ O, const bf16_t* __restrict__ E,
    const float* __restrict__ obias,
    float* __restrict__ pmax, float* __restrict__ psum) {
  int wid = threadIdx.x >> 5;
  int wm = wid & 3, wn = wid >> 2;
  int lane = threadIdx.x & 31;
  int halfsel = lane >> 4;
  int rowBlock = blockIdx.x * 64;
  int row0 = rowBlock + wm * 16;
  int colBase = blockIdx.y * CHC;

  v16bf afr[16];
  const bf16_t* Arow = O + (long)row0 * FD;
  for (int kk = 0; kk < 16; ++kk) afr[kk] = load_frag(Arow + kk * 32, FD);

  float mrun[8], srun[8];
  for (int r = 0; r < 8; ++r) { mrun[r] = -3.0e38f; srun[r] = 0.f; }

  for (int it = 0; it < CHC / 32; ++it) {
    int c0 = colBase + it * 32 + wn * 16;
    v8f c{};
    const bf16_t* Brow = E + (long)c0 * FD;
    for (int kk = 0; kk < 16; ++kk)
      c = wmma_bf16(afr[kk], load_frag(Brow + kk * 32, FD), c);
    float bias = obias[c0 + (lane & 15)];
    for (int r = 0; r < 8; ++r) {
      float l = c[r] + bias;
      float tm = l;
      tm = fmaxf(tm, __shfl_xor(tm, 1, 32));
      tm = fmaxf(tm, __shfl_xor(tm, 2, 32));
      tm = fmaxf(tm, __shfl_xor(tm, 4, 32));
      tm = fmaxf(tm, __shfl_xor(tm, 8, 32));
      float mn = fmaxf(mrun[r], tm);
      float p = __expf(l - mn);
      p += __shfl_xor(p, 1, 32);
      p += __shfl_xor(p, 2, 32);
      p += __shfl_xor(p, 4, 32);
      p += __shfl_xor(p, 8, 32);
      srun[r] = srun[r] * __expf(mrun[r] - mn) + p;
      mrun[r] = mn;
    }
  }
  __shared__ float smax[2][64];
  __shared__ float ssum[2][64];
  if ((lane & 15) == 0) {
    for (int r = 0; r < 8; ++r) {
      int rr = wm * 16 + r + halfsel * 8;
      smax[wn][rr] = mrun[r];
      ssum[wn][rr] = srun[r];
    }
  }
  __syncthreads();
  if (threadIdx.x < 64) {
    float m0 = smax[0][threadIdx.x], m1 = smax[1][threadIdx.x];
    float m = fmaxf(m0, m1);
    float s = ssum[0][threadIdx.x] * __expf(m0 - m) +
              ssum[1][threadIdx.x] * __expf(m1 - m);
    int row = rowBlock + (int)threadIdx.x;
    pmax[(long)row * NCH + blockIdx.y] = m;
    psum[(long)row * NCH + blockIdx.y] = s;
  }
}

__global__ void stats_reduce_kernel(const float* __restrict__ pmax,
                                    const float* __restrict__ psum,
                                    float* __restrict__ rmax,
                                    float* __restrict__ rsum) {
  int row = blockIdx.x * blockDim.x + threadIdx.x;
  if (row >= NT) return;
  float m = -3.0e38f;
  for (int c = 0; c < NCH; ++c) m = fmaxf(m, pmax[(long)row * NCH + c]);
  float s = 0.f;
  for (int c = 0; c < NCH; ++c)
    s += psum[(long)row * NCH + c] * __expf(pmax[(long)row * NCH + c] - m);
  rmax[row] = m;
  rsum[row] = s;
}

// ---------------------------------------------------------------------------
// Recompute logits chunk, P = softmax - onehot (bf16). grid=(NT/64, CHB/CPW).
// ---------------------------------------------------------------------------
__global__ __launch_bounds__(256) void compute_p_kernel(
    const bf16_t* __restrict__ O, const bf16_t* __restrict__ E,
    const float* __restrict__ obias, const float* __restrict__ rmax,
    const float* __restrict__ rsum, const int* __restrict__ labels,
    bf16_t* __restrict__ P, int col0) {
  int wid = threadIdx.x >> 5;
  int wm = wid & 3, wn = wid >> 2;
  int lane = threadIdx.x & 31;
  int halfsel = lane >> 4;
  int row0 = blockIdx.x * 64 + wm * 16;
  int cstart = col0 + blockIdx.y * CPW;

  v16bf afr[16];
  const bf16_t* Arow = O + (long)row0 * FD;
  for (int kk = 0; kk < 16; ++kk) afr[kk] = load_frag(Arow + kk * 32, FD);

  float rm[8], ri[8];
  int lab[8];
  for (int r = 0; r < 8; ++r) {
    int row = row0 + r + halfsel * 8;
    rm[r] = rmax[row];
    ri[r] = 1.f / rsum[row];
    lab[r] = labels[row];
  }
  for (int it = 0; it < CPW / 32; ++it) {
    int c0 = cstart + it * 32 + wn * 16;
    v8f c{};
    const bf16_t* Brow = E + (long)c0 * FD;
    for (int kk = 0; kk < 16; ++kk)
      c = wmma_bf16(afr[kk], load_frag(Brow + kk * 32, FD), c);
    int col = c0 + (lane & 15);
    float bias = obias[col];
    for (int r = 0; r < 8; ++r) {
      float p = __expf(c[r] + bias - rm[r]) * ri[r];
      if (col == lab[r]) p -= 1.f;
      int row = row0 + r + halfsel * 8;
      P[(long)row * CHB + (col - col0)] = f2bf(p);
    }
  }
}

// ---------------------------------------------------------------------------
// LayerNorm forward / backward (one block of 256 per row).
// ---------------------------------------------------------------------------
__global__ __launch_bounds__(256) void ln_fwd_kernel(
    const float* __restrict__ X, const float* __restrict__ gamma,
    const float* __restrict__ beta, float* __restrict__ mu_o,
    float* __restrict__ rs_o, float* __restrict__ Yf,
    bf16_t* __restrict__ Yb) {
  int row = blockIdx.x;
  const float* x = X + (long)row * FD;
  int tid = threadIdx.x;
  float s = 0.f, s2 = 0.f;
  for (int f = tid; f < FD; f += 256) { float v = x[f]; s += v; s2 += v * v; }
  __shared__ float red[2][8];
  for (int m = 16; m; m >>= 1) {
    s += __shfl_xor(s, m, 32);
    s2 += __shfl_xor(s2, m, 32);
  }
  int wid = tid >> 5, lane = tid & 31;
  if (lane == 0) { red[0][wid] = s; red[1][wid] = s2; }
  __syncthreads();
  if (tid == 0) {
    float a = 0.f, b = 0.f;
    for (int i = 0; i < 8; ++i) { a += red[0][i]; b += red[1][i]; }
    red[0][0] = a; red[1][0] = b;
  }
  __syncthreads();
  float mu = red[0][0] / FD;
  float var = red[1][0] / FD - mu * mu;
  float rs = rsqrtf(var + 1e-5f);
  if (tid == 0 && mu_o) { mu_o[row] = mu; rs_o[row] = rs; }
  for (int f = tid; f < FD; f += 256) {
    float y = (x[f] - mu) * rs * gamma[f] + beta[f];
    if (Yf) Yf[(long)row * FD + f] = y;
    Yb[(long)row * FD + f] = f2bf(y);
  }
}

__global__ __launch_bounds__(256) void ln_bwd_kernel(
    const float* __restrict__ dY, const float* __restrict__ X,
    const float* __restrict__ mu_a, const float* __restrict__ rs_a,
    const float* __restrict__ gamma, float* __restrict__ dX) {
  int row = blockIdx.x;
  float mu = mu_a[row], rs = rs_a[row];
  const float* x = X + (long)row * FD;
  const float* dy = dY + (long)row * FD;
  int tid = threadIdx.x;
  float sg = 0.f, sgz = 0.f;
  for (int f = tid; f < FD; f += 256) {
    float g = dy[f] * gamma[f];
    float z = (x[f] - mu) * rs;
    sg += g; sgz += g * z;
  }
  __shared__ float red[2][8];
  for (int m = 16; m; m >>= 1) {
    sg += __shfl_xor(sg, m, 32);
    sgz += __shfl_xor(sgz, m, 32);
  }
  int wid = tid >> 5, lane = tid & 31;
  if (lane == 0) { red[0][wid] = sg; red[1][wid] = sgz; }
  __syncthreads();
  if (tid == 0) {
    float a = 0.f, b = 0.f;
    for (int i = 0; i < 8; ++i) { a += red[0][i]; b += red[1][i]; }
    red[0][0] = a / FD; red[1][0] = b / FD;
  }
  __syncthreads();
  float mg = red[0][0], mgz = red[1][0];
  for (int f = tid; f < FD; f += 256) {
    float g = dy[f] * gamma[f];
    float z = (x[f] - mu) * rs;
    dX[(long)row * FD + f] = rs * (g - mg - z * mgz);
  }
}

// ---------------------------------------------------------------------------
// Small utility kernels
// ---------------------------------------------------------------------------
__global__ void zero_kernel(float* __restrict__ p, long n) {
  long i = (long)blockIdx.x * blockDim.x + threadIdx.x;
  if (i < n) p[i] = 0.f;
}

__global__ void cvt_bf_kernel(const float* __restrict__ src,
                              bf16_t* __restrict__ dst, long n) {
  long i = (long)blockIdx.x * blockDim.x + threadIdx.x;
  if (i < n) dst[i] = f2bf(src[i]);
}

// dst[c*R + r] = bf16(src[r*C + c]); grid=(C/32, R/32), block=256 (32x8)
__global__ __launch_bounds__(256) void cvt_tr_kernel(
    const float* __restrict__ src, bf16_t* __restrict__ dst, int R, int C) {
  __shared__ float tile[32][33];
  int tx = threadIdx.x & 31, ty = threadIdx.x >> 5;
  int c0 = blockIdx.x * 32, r0 = blockIdx.y * 32;
  for (int i = 0; i < 32; i += 8)
    tile[ty + i][tx] = src[(long)(r0 + ty + i) * C + c0 + tx];
  __syncthreads();
  for (int i = 0; i < 32; i += 8)
    dst[(long)(c0 + ty + i) * R + r0 + tx] = f2bf(tile[tx][ty + i]);
}

// out[f] += sum over rows chunk of X[:,f]; grid=(F/256, N/rowsPer)
__global__ void colsum_kernel(const float* __restrict__ X,
                              float* __restrict__ out, int F, int rowsPer) {
  int f = blockIdx.x * blockDim.x + threadIdx.x;
  int r0 = blockIdx.y * rowsPer;
  float s = 0.f;
  for (int r = 0; r < rowsPer; ++r) s += X[(long)(r0 + r) * F + f];
  atomicAdd(&out[f], s);
}

__global__ void relu_mask_kernel(const float* __restrict__ dv,
                                 const float* __restrict__ v,
                                 float* __restrict__ dvr, long n) {
  long i = (long)blockIdx.x * blockDim.x + threadIdx.x;
  if (i < n) dvr[i] = (v[i] > 0.f) ? dv[i] : 0.f;
}

__global__ void mul_cvt_kernel(const float* __restrict__ a,
                               const float* __restrict__ b,
                               bf16_t* __restrict__ o, long n) {
  long i = (long)blockIdx.x * blockDim.x + threadIdx.x;
  if (i < n) o[i] = f2bf(a[i] * b[i]);
}

// ofp = opre - v*t - step_b[f]*v*s[row]
__global__ void ofast_pre_kernel(const float* __restrict__ opre,
                                 const float* __restrict__ v,
                                 const float* __restrict__ t,
                                 const float* __restrict__ s,
                                 const float* __restrict__ step_b,
                                 float* __restrict__ o, long n) {
  long i = (long)blockIdx.x * blockDim.x + threadIdx.x;
  if (i < n) {
    int f = (int)(i % FD);
    long row = i / FD;
    o[i] = opre[i] - v[i] * t[i] - step_b[f] * v[i] * s[row];
  }
}

// one wave per row: out[row] = dot(V[row,:], g)
__global__ void rowdot_kernel(const float* __restrict__ V,
                              const float* __restrict__ g,
                              float* __restrict__ out) {
  int wid = threadIdx.x >> 5, lane = threadIdx.x & 31;
  int row = blockIdx.x * 8 + wid;
  const float* x = V + (long)row * FD;
  float s = 0.f;
  for (int f = lane; f < FD; f += 32) s += x[f] * g[f];
  for (int m = 16; m; m >>= 1) s += __shfl_xor(s, m, 32);
  if (lane == 0) out[row] = s;
}

__global__ void label_logit_kernel(const float* __restrict__ O,
                                   const float* __restrict__ E,
                                   const float* __restrict__ obias,
                                   const int* __restrict__ labels,
                                   float* __restrict__ out) {
  int wid = threadIdx.x >> 5, lane = threadIdx.x & 31;
  int row = blockIdx.x * 8 + wid;
  int lb = labels[row];
  const float* x = O + (long)row * FD;
  const float* e = E + (long)lb * FD;
  float s = 0.f;
  for (int f = lane; f < FD; f += 32) s += x[f] * e[f];
  for (int m = 16; m; m >>= 1) s += __shfl_xor(s, m, 32);
  if (lane == 0) out[row] = s + obias[lb];
}

__global__ void fast_loss_kernel(const float* __restrict__ rmax,
                                 const float* __restrict__ rsum,
                                 const float* __restrict__ ll,
                                 float* __restrict__ out) {
  int i = blockIdx.x * blockDim.x + threadIdx.x;
  if (i < NT) out[i] = rmax[i] + logf(rsum[i]) - ll[i];
}

// ---------------------------------------------------------------------------
// Host side
// ---------------------------------------------------------------------------
static inline char* carve(char*& p, size_t bytes) {
  char* r = p;
  p += (bytes + 255) & ~(size_t)255;
  return r;
}

extern "C" void kernel_launch(void* const* d_in, const int* in_sizes, int n_in,
                              void* d_out, int out_size, void* d_ws,
                              size_t ws_size, hipStream_t stream) {
  (void)in_sizes; (void)n_in; (void)out_size; (void)ws_size;
  const float* h        = (const float*)d_in[0];
  const float* Ef       = (const float*)d_in[1];
  const float* obias    = (const float*)d_in[2];
  const int*   labels   = (const int*)d_in[3];
  const float* Wh       = (const float*)d_in[4];
  const float* b_hidden = (const float*)d_in[5];
  const float* hid_bias = (const float*)d_in[6];
  const float* Wp       = (const float*)d_in[7];
  const float* bp       = (const float*)d_in[8];
  const float* gamma    = (const float*)d_in[9];
  const float* beta     = (const float*)d_in[10];
  const float* stepW    = (const float*)d_in[11];
  const float* stepb    = (const float*)d_in[12];

  float* out       = (float*)d_out;
  float* out_loss  = out;                       // [NT]
  float* out_gWh   = out + NT;                  // [FD,HD]
  float* out_ghb   = out_gWh + (long)FD * HD;   // [FD]
  float* out_gWp   = out_ghb + FD;              // [FD,FD]
  float* out_gbp   = out_gWp + (long)FD * FD;   // [FD]

  char* p = (char*)d_ws;
  bf16_t* h_bf    = (bf16_t*)carve(p, (size_t)NT * HD * 2);
  bf16_t* hT_bf   = (bf16_t*)carve(p, (size_t)HD * NT * 2);
  bf16_t* Wh_bf   = (bf16_t*)carve(p, (size_t)FD * HD * 2);
  bf16_t* Wp_bf   = (bf16_t*)carve(p, (size_t)FD * FD * 2);
  bf16_t* WpT_bf  = (bf16_t*)carve(p, (size_t)FD * FD * 2);
  bf16_t* sW_bf   = (bf16_t*)carve(p, (size_t)FD * FD * 2);
  bf16_t* E_bf    = (bf16_t*)carve(p, (size_t)VD * FD * 2);
  bf16_t* ET_bf   = (bf16_t*)carve(p, (size_t)FD * VD * 2);
  float*  v_f     = (float*)carve(p, (size_t)NT * FD * 4);
  bf16_t* v_bf    = (bf16_t*)carve(p, (size_t)NT * FD * 2);
  bf16_t* vT_bf   = (bf16_t*)carve(p, (size_t)FD * NT * 2);
  float*  opre    = (float*)carve(p, (size_t)NT * FD * 4);
  float*  mu      = (float*)carve(p, (size_t)NT * 4);
  float*  rsig    = (float*)carve(p, (size_t)NT * 4);
  bf16_t* o_bf    = (bf16_t*)carve(p, (size_t)NT * FD * 2);
  float*  pmax    = (float*)carve(p, (size_t)NT * NCH * 4);
  float*  psum    = (float*)carve(p, (size_t)NT * NCH * 4);
  float*  rmax    = (float*)carve(p, (size_t)NT * 4);
  float*  rsum    = (float*)carve(p, (size_t)NT * 4);
  bf16_t* Pch     = (bf16_t*)carve(p, (size_t)NT * CHB * 2);
  float*  dO      = (float*)carve(p, (size_t)NT * FD * 4);
  float*  dop     = (float*)carve(p, (size_t)NT * FD * 4);
  bf16_t* dop_bf  = (bf16_t*)carve(p, (size_t)NT * FD * 2);
  bf16_t* dopT_bf = (bf16_t*)carve(p, (size_t)FD * NT * 2);
  float*  dv      = (float*)carve(p, (size_t)NT * FD * 4);
  float*  dvr     = (float*)carve(p, (size_t)NT * FD * 4);
  bf16_t* dvrT_bf = (bf16_t*)carve(p, (size_t)FD * NT * 2);
  bf16_t* gWpT_bf = (bf16_t*)carve(p, (size_t)FD * FD * 2);
  float*  u_f     = (float*)carve(p, (size_t)NT * FD * 4);
  bf16_t* w_bf    = (bf16_t*)carve(p, (size_t)NT * FD * 2);
  float*  t_f     = (float*)carve(p, (size_t)NT * FD * 4);
  float*  s_tok   = (float*)carve(p, (size_t)NT * 4);
  float*  ofp     = (float*)carve(p, (size_t)NT * FD * 4);
  float*  ofast_f = (float*)carve(p, (size_t)NT * FD * 4);
  bf16_t* ofast_bf= (bf16_t*)carve(p, (size_t)NT * FD * 2);
  float*  pmaxf   = (float*)carve(p, (size_t)NT * NCH * 4);
  float*  psumf   = (float*)carve(p, (size_t)NT * NCH * 4);
  float*  rmaxf   = (float*)carve(p, (size_t)NT * 4);
  float*  rsumf   = (float*)carve(p, (size_t)NT * 4);
  float*  ll      = (float*)carve(p, (size_t)NT * 4);

  const dim3 B256(256);
#define GRID1(n) dim3((unsigned)(((long)(n) + 255) / 256))

  // ---- conversions -------------------------------------------------------
  cvt_bf_kernel<<<GRID1((long)NT * HD), B256, 0, stream>>>(h, h_bf, (long)NT * HD);
  cvt_bf_kernel<<<GRID1((long)FD * HD), B256, 0, stream>>>(Wh, Wh_bf, (long)FD * HD);
  cvt_bf_kernel<<<GRID1((long)FD * FD), B256, 0, stream>>>(Wp, Wp_bf, (long)FD * FD);
  cvt_bf_kernel<<<GRID1((long)FD * FD), B256, 0, stream>>>(stepW, sW_bf, (long)FD * FD);
  cvt_bf_kernel<<<GRID1((long)VD * FD), B256, 0, stream>>>(Ef, E_bf, (long)VD * FD);
  cvt_tr_kernel<<<dim3(HD / 32, NT / 32), B256, 0, stream>>>(h, hT_bf, NT, HD);
  cvt_tr_kernel<<<dim3(FD / 32, FD / 32), B256, 0, stream>>>(Wp, WpT_bf, FD, FD);
  cvt_tr_kernel<<<dim3(FD / 32, VD / 32), B256, 0, stream>>>(Ef, ET_bf, VD, FD);

  // ---- slow forward ------------------------------------------------------
  // v = relu(h @ Wh.T + b_hidden + hidden_bias)
  gemm_bt_kernel<<<dim3(NT / 128, FD / 64), B256, 0, stream>>>(
      h_bf, HD, Wh_bf, HD, v_f, FD, HD, b_hidden, hid_bias, 1, 0, v_bf, FD);
  cvt_tr_kernel<<<dim3(FD / 32, NT / 32), B256, 0, stream>>>(v_f, vT_bf, NT, FD);
  // o_pre = v @ Wp.T + bp
  gemm_bt_kernel<<<dim3(NT / 128, FD / 64), B256, 0, stream>>>(
      v_bf, FD, Wp_bf, FD, opre, FD, FD, bp, nullptr, 0, 0, nullptr, 0);
  ln_fwd_kernel<<<dim3(NT), B256, 0, stream>>>(opre, gamma, beta, mu, rsig,
                                               nullptr, o_bf);
  // slow softmax stats
  logits_stats_kernel<<<dim3(NT / 64, NCH), B256, 0, stream>>>(o_bf, E_bf, obias,
                                                               pmax, psum);
  stats_reduce_kernel<<<GRID1(NT), B256, 0, stream>>>(pmax, psum, rmax, rsum);

  // ---- slow backward: dO = (softmax - onehot) @ E, chunked over V --------
  zero_kernel<<<GRID1((long)NT * FD), B256, 0, stream>>>(dO, (long)NT * FD);
  for (int c = 0; c < NPB; ++c) {
    compute_p_kernel<<<dim3(NT / 64, CHB / CPW), B256, 0, stream>>>(
        o_bf, E_bf, obias, rmax, rsum, labels, Pch, c * CHB);
    gemm_bt_kernel<<<dim3(NT / 128, FD / 64), B256, 0, stream>>>(
        Pch, CHB, ET_bf + (long)c * CHB, VD, dO, FD, CHB, nullptr, nullptr, 0,
        1, nullptr, 0);
  }
  // LN backward -> dop
  ln_bwd_kernel<<<dim3(NT), B256, 0, stream>>>(dO, opre, mu, rsig, gamma, dop);
  // gbp = colsum(dop)
  zero_kernel<<<GRID1(FD), B256, 0, stream>>>(out_gbp, FD);
  colsum_kernel<<<dim3(FD / 256, NT / 128), B256, 0, stream>>>(dop, out_gbp, FD, 128);
  cvt_tr_kernel<<<dim3(FD / 32, NT / 32), B256, 0, stream>>>(dop, dopT_bf, NT, FD);
  cvt_bf_kernel<<<GRID1((long)NT * FD), B256, 0, stream>>>(dop, dop_bf, (long)NT * FD);
  // gWp = dop.T @ v
  gemm_bt_kernel<<<dim3(FD / 128, FD / 64), B256, 0, stream>>>(
      dopT_bf, NT, vT_bf, NT, out_gWp, FD, NT, nullptr, nullptr, 0, 0, nullptr, 0);
  // dv = dop @ Wp ; relu mask
  gemm_bt_kernel<<<dim3(NT / 128, FD / 64), B256, 0, stream>>>(
      dop_bf, FD, WpT_bf, FD, dv, FD, FD, nullptr, nullptr, 0, 0, nullptr, 0);
  relu_mask_kernel<<<GRID1((long)NT * FD), B256, 0, stream>>>(dv, v_f, dvr,
                                                              (long)NT * FD);
  zero_kernel<<<GRID1(FD), B256, 0, stream>>>(out_ghb, FD);
  colsum_kernel<<<dim3(FD / 256, NT / 128), B256, 0, stream>>>(dvr, out_ghb, FD, 128);
  cvt_tr_kernel<<<dim3(FD / 32, NT / 32), B256, 0, stream>>>(dvr, dvrT_bf, NT, FD);
  // gWh = dvr.T @ h
  gemm_bt_kernel<<<dim3(FD / 128, HD / 64), B256, 0, stream>>>(
      dvrT_bf, NT, hT_bf, NT, out_gWh, HD, NT, nullptr, nullptr, 0, 0, nullptr, 0);

  // ---- fast-weight path --------------------------------------------------
  cvt_tr_kernel<<<dim3(FD / 32, FD / 32), B256, 0, stream>>>(out_gWp, gWpT_bf, FD, FD);
  // u = v @ gWp
  gemm_bt_kernel<<<dim3(NT / 128, FD / 64), B256, 0, stream>>>(
      v_bf, FD, gWpT_bf, FD, u_f, FD, FD, nullptr, nullptr, 0, 0, nullptr, 0);
  // w = v*u ; t = w @ step_W.T
  mul_cvt_kernel<<<GRID1((long)NT * FD), B256, 0, stream>>>(v_f, u_f, w_bf,
                                                            (long)NT * FD);
  gemm_bt_kernel<<<dim3(NT / 128, FD / 64), B256, 0, stream>>>(
      w_bf, FD, sW_bf, FD, t_f, FD, FD, nullptr, nullptr, 0, 0, nullptr, 0);
  // s = v . gbp
  rowdot_kernel<<<dim3(NT / 8), B256, 0, stream>>>(v_f, out_gbp, s_tok);
  ofast_pre_kernel<<<GRID1((long)NT * FD), B256, 0, stream>>>(
      opre, v_f, t_f, s_tok, stepb, ofp, (long)NT * FD);
  ln_fwd_kernel<<<dim3(NT), B256, 0, stream>>>(ofp, gamma, beta, nullptr,
                                               nullptr, ofast_f, ofast_bf);
  // fast softmax stats + label logit + loss
  logits_stats_kernel<<<dim3(NT / 64, NCH), B256, 0, stream>>>(
      ofast_bf, E_bf, obias, pmaxf, psumf);
  stats_reduce_kernel<<<GRID1(NT), B256, 0, stream>>>(pmaxf, psumf, rmaxf, rsumf);
  label_logit_kernel<<<dim3(NT / 8), B256, 0, stream>>>(ofast_f, Ef, obias,
                                                        labels, ll);
  fast_loss_kernel<<<GRID1(NT), B256, 0, stream>>>(rmaxf, rsumf, ll, out_loss);
#undef GRID1
}
